// GNN_27891517620521
// MI455X (gfx1250) — compile-verified, matrix operand order checked by reading, not run on hardware
//
#include <hip/hip_runtime.h>

// ---------------------------------------------------------------------------
// GraphSAGE (3x SAGEConv mean + MLP head) for MI455X / gfx1250.
//
// Perf model: GEMMs total ~15 GFLOP (noise at WMMA rates). The edge scatter
// dominates: ~3 GB of irregular traffic, but N*256 f32 features (51 MB) +
// agg buffer fit in the 192 MB L2, so f32 global atomics stay L2-resident.
// Dense math: f32->f16 staged in LDS, v_wmma_f32_16x16x32_f16, f32 accum.
// ---------------------------------------------------------------------------

typedef _Float16 v8h  __attribute__((ext_vector_type(8)));
typedef _Float16 v16h __attribute__((ext_vector_type(16)));
typedef float    v8f  __attribute__((ext_vector_type(8)));

// ---------------------------------------------------------------------------
// Utility kernels
// ---------------------------------------------------------------------------
__global__ void gnn_zero_f32v4(float4* __restrict__ p, int n4) {
  int i = blockIdx.x * blockDim.x + threadIdx.x;
  if (i < n4) p[i] = make_float4(0.f, 0.f, 0.f, 0.f);
}

__global__ void gnn_degree(const long long* __restrict__ tgt,
                           float* __restrict__ deg, int E) {
  int e = blockIdx.x * blockDim.x + threadIdx.x;
  if (e < E) atomicAdd(&deg[tgt[e]], 1.0f);
}

__global__ void gnn_invdeg(const float* __restrict__ deg,
                           float* __restrict__ invdeg, int n) {
  int i = blockIdx.x * blockDim.x + threadIdx.x;
  if (i < n) invdeg[i] = 1.0f / fmaxf(deg[i], 1.0f);
}

// Edge scatter: agg[tgt] += feat[src], 4 floats per thread.
// perShift = log2(Din/4); consecutive threads cover one edge's features
// (coalesced row gather), then the next edge.
__global__ void gnn_scatter(const float* __restrict__ feat,
                            const long long* __restrict__ src,
                            const long long* __restrict__ tgt,
                            float* __restrict__ agg, int E, int Din,
                            int perShift) {
  int idx = blockIdx.x * blockDim.x + threadIdx.x;
  int e = idx >> perShift;
  if (e >= E) return;
  int f = (idx & ((1 << perShift) - 1)) << 2;
  long long s = src[e];
  long long t = tgt[e];
  const float4 v = *(const float4*)(feat + (size_t)s * Din + f);
  float* d = agg + (size_t)t * Din + f;
  atomicAdd(d + 0, v.x);
  atomicAdd(d + 1, v.y);
  atomicAdd(d + 2, v.z);
  atomicAdd(d + 3, v.w);
}

// ---------------------------------------------------------------------------
// Fused SAGE GEMM:
//   out[M,Nout] = act( (A1 * scale[row]) @ W1  +  A2 @ W2  + bias )
// as one GEMM with K = K1 + K2 (A2/W2 optional -> K2 = 0; K1,K2 % 64 == 0).
// Tile: 128(M) x 64(N) per 256-thread block; 8 wave32s in a 4x2 grid; each
// wave owns 32x32 as 2x2 v_wmma_f32_16x16x32_f16 fragments; BK = 32.
// ---------------------------------------------------------------------------
#define BM 128
#define BN 64
#define BK 32
#define LDSK 40   // padded K stride (elements) -> 80B rows, 16B aligned

__global__ void __launch_bounds__(256)
gnn_sage_gemm(const float* __restrict__ A1, const float* __restrict__ scale,
              int K1,
              const float* __restrict__ A2, int K2,
              const float* __restrict__ W1, const float* __restrict__ W2,
              const float* __restrict__ bias, float* __restrict__ out,
              int M, int Nout, int relu) {
  __shared__ __align__(16) _Float16 lds_a[BM][LDSK];
  __shared__ __align__(16) _Float16 lds_b[BN][LDSK];

  const int tid  = threadIdx.x;
  const int lane = tid & 31;
  const int wave = tid >> 5;        // 0..7
  const int waveM = wave & 3;       // 4 waves along M
  const int waveN = wave >> 2;      // 2 waves along N
  const int l16   = lane & 15;
  const int khalf = (lane >> 4) << 3;   // 0 or 8 (ISA 16-bit A/B layout)

  const int rowBase = blockIdx.x * BM;
  const int colBase = blockIdx.y * BN;
  const int Ktot = K1 + K2;

  // Per-thread staging roles (constant across K loop)
  const int aRow   = tid >> 1;           // 0..127
  const int aKpart = (tid & 1) << 4;     // 0 or 16
  const int aGrow  = rowBase + aRow;
  const bool aIn   = (aGrow < M);
  const float aScl = (scale && aIn) ? scale[aGrow] : 1.0f;

  const int bN    = tid & 63;            // 0..63
  const int bKgrp = (tid >> 6) << 3;     // 0,8,16,24
  const int bCol  = colBase + bN;
  const bool bIn  = (bCol < Nout);

  v8f acc[2][2];
  #pragma unroll
  for (int mi = 0; mi < 2; ++mi)
    #pragma unroll
    for (int ni = 0; ni < 2; ++ni)
      acc[mi][ni] = (v8f)(0.0f);

  for (int k0 = 0; k0 < Ktot; k0 += BK) {
    // ---- stage A tile: 128 rows x 32 k, f32 -> f16, packed b128 stores ----
    {
      v8h h0 = {};
      v8h h1 = {};
      if (aIn) {
        const int kg0 = k0 + aKpart;     // 16-chunk never straddles K1
        const float* srcp;
        float s;
        if (kg0 < K1) { srcp = A1 + (size_t)aGrow * K1 + kg0;        s = aScl; }
        else          { srcp = A2 + (size_t)aGrow * K2 + (kg0 - K1); s = 1.0f; }
        float4 v0 = ((const float4*)srcp)[0];
        float4 v1 = ((const float4*)srcp)[1];
        float4 v2 = ((const float4*)srcp)[2];
        float4 v3 = ((const float4*)srcp)[3];
        __builtin_prefetch(srcp + BK, 0, 1);   // next K-slice -> global_prefetch
        h0[0] = (_Float16)(v0.x * s); h0[1] = (_Float16)(v0.y * s);
        h0[2] = (_Float16)(v0.z * s); h0[3] = (_Float16)(v0.w * s);
        h0[4] = (_Float16)(v1.x * s); h0[5] = (_Float16)(v1.y * s);
        h0[6] = (_Float16)(v1.z * s); h0[7] = (_Float16)(v1.w * s);
        h1[0] = (_Float16)(v2.x * s); h1[1] = (_Float16)(v2.y * s);
        h1[2] = (_Float16)(v2.z * s); h1[3] = (_Float16)(v2.w * s);
        h1[4] = (_Float16)(v3.x * s); h1[5] = (_Float16)(v3.y * s);
        h1[6] = (_Float16)(v3.z * s); h1[7] = (_Float16)(v3.w * s);
      }
      *(v8h*)&lds_a[aRow][aKpart]     = h0;
      *(v8h*)&lds_a[aRow][aKpart + 8] = h1;
    }
    // ---- stage B tile transposed: lds_b[n][k], one b128 store/thread ----
    {
      v8h hb = {};
      if (bIn) {
        const int kb = k0 + bKgrp;       // 8-chunk never straddles K1
        const float* wp = (kb < K1) ? (W1 + (size_t)kb * Nout + bCol)
                                    : (W2 + (size_t)(kb - K1) * Nout + bCol);
        #pragma unroll
        for (int j = 0; j < 8; ++j)
          hb[j] = (_Float16)wp[(size_t)j * Nout];
      }
      *(v8h*)&lds_b[bN][bKgrp] = hb;
    }
    __syncthreads();

    // ---- fragments (lanes 0-15: K, K+16 ; lanes 16-31: K+8, K+24) ----
    v16h aF[2], bF[2];
    #pragma unroll
    for (int mi = 0; mi < 2; ++mi) {
      const int r = waveM * 32 + mi * 16 + l16;
      v8h lo = *(const v8h*)&lds_a[r][khalf];
      v8h hi = *(const v8h*)&lds_a[r][16 + khalf];
      aF[mi] = __builtin_shufflevector(lo, hi, 0,1,2,3,4,5,6,7,8,9,10,11,12,13,14,15);
    }
    #pragma unroll
    for (int ni = 0; ni < 2; ++ni) {
      const int n = waveN * 32 + ni * 16 + l16;
      v8h lo = *(const v8h*)&lds_b[n][khalf];
      v8h hi = *(const v8h*)&lds_b[n][16 + khalf];
      bF[ni] = __builtin_shufflevector(lo, hi, 0,1,2,3,4,5,6,7,8,9,10,11,12,13,14,15);
    }

    #pragma unroll
    for (int mi = 0; mi < 2; ++mi)
      #pragma unroll
      for (int ni = 0; ni < 2; ++ni)
        acc[mi][ni] = __builtin_amdgcn_wmma_f32_16x16x32_f16(
            false, aF[mi], false, bF[ni], (short)0, acc[mi][ni], false, false);

    __syncthreads();
  }

  // ---- epilogue: bias + optional relu ----
  const bool fullM = (rowBase + BM <= M);
  #pragma unroll
  for (int ni = 0; ni < 2; ++ni) {
    const int col = colBase + waveN * 32 + ni * 16 + l16;
    if (col >= Nout) continue;
    const float bv = bias[col];
    #pragma unroll
    for (int mi = 0; mi < 2; ++mi) {
      const int rOff = rowBase + waveM * 32 + mi * 16 + ((lane >> 4) << 3);
      if (fullM) {
        float* op = out + (size_t)rOff * Nout + col;
        #pragma unroll
        for (int r = 0; r < 8; ++r) {
          float v = acc[mi][ni][r] + bv;
          if (relu) v = fmaxf(v, 0.0f);
          op[(size_t)r * Nout] = v;
        }
      } else {
        #pragma unroll
        for (int r = 0; r < 8; ++r) {
          const int row = rOff + r;
          if (row < M) {
            float v = acc[mi][ni][r] + bv;
            if (relu) v = fmaxf(v, 0.0f);
            out[(size_t)row * Nout + col] = v;
          }
        }
      }
    }
  }
}

// ---------------------------------------------------------------------------
// Host-side orchestration
// ---------------------------------------------------------------------------
static inline void launch_zero(float* p, long long n, hipStream_t s) {
  long long n4 = n >> 2;   // all our sizes are multiples of 4
  gnn_zero_f32v4<<<(unsigned)((n4 + 255) / 256), 256, 0, s>>>((float4*)p, (int)n4);
}

static inline void launch_scatter(const float* feat, const long long* src,
                                  const long long* tgt, float* agg,
                                  int E, int Din, hipStream_t s) {
  int per = Din >> 2;
  int shift = __builtin_ctz(per);
  long long tot = (long long)E * per;
  gnn_scatter<<<(unsigned)((tot + 255) / 256), 256, 0, s>>>(feat, src, tgt, agg,
                                                            E, Din, shift);
}

static inline void launch_gemm(const float* A1, const float* scale, int K1,
                               const float* A2, int K2,
                               const float* W1, const float* W2,
                               const float* bias, float* out,
                               int M, int Nout, int relu, hipStream_t s) {
  dim3 grid((M + BM - 1) / BM, (Nout + BN - 1) / BN);
  gnn_sage_gemm<<<grid, 256, 0, s>>>(A1, scale, K1, A2, K2, W1, W2, bias, out,
                                     M, Nout, relu);
}

extern "C" void kernel_launch(void* const* d_in, const int* in_sizes, int n_in,
                              void* d_out, int out_size, void* d_ws, size_t ws_size,
                              hipStream_t stream) {
  const float*     x      = (const float*)d_in[0];
  const long long* edges  = (const long long*)d_in[1];   // int64 per reference
  const float* Wf_l   = (const float*)d_in[2];
  const float* bf     = (const float*)d_in[3];
  const float* Wf_r   = (const float*)d_in[4];
  const float* W0_l   = (const float*)d_in[5];
  const float* b0     = (const float*)d_in[6];
  const float* W0_r   = (const float*)d_in[7];
  const float* W1_l   = (const float*)d_in[8];
  const float* b1     = (const float*)d_in[9];
  const float* W1_r   = (const float*)d_in[10];
  const float* W_lin  = (const float*)d_in[11];
  const float* b_lin  = (const float*)d_in[12];
  const float* W_lin2 = (const float*)d_in[13];
  const float* b_lin2 = (const float*)d_in[14];

  const int N = in_sizes[0] / 64;     // 50000
  const int E = in_sizes[1] / 2;      // 800000
  const long long* src = edges;
  const long long* tgt = edges + E;

  // Workspace layout (floats): 2N + 3*N*256  (~154 MB)
  float* ws     = (float*)d_ws;
  float* deg    = ws;                 ws += N;
  float* invdeg = ws;                 ws += N;
  float* agg    = ws;                 ws += (size_t)N * 256;
  float* bufA   = ws;                 ws += (size_t)N * 256;
  float* bufB   = ws;                 // N*256

  // Degrees (recomputed each call -> deterministic)
  launch_zero(deg, N, stream);
  gnn_degree<<<(E + 255) / 256, 256, 0, stream>>>(tgt, deg, E);
  gnn_invdeg<<<(N + 255) / 256, 256, 0, stream>>>(deg, invdeg, N);

  // ---- Layer 1: 64 -> 256 ----
  launch_zero(agg, (long long)N * 64, stream);
  launch_scatter(x, src, tgt, agg, E, 64, stream);
  launch_gemm(agg, invdeg, 64, x, 64, Wf_l, Wf_r, bf, bufA, N, 256, 1, stream);

  // ---- Layer 2: 256 -> 128 ----
  launch_zero(agg, (long long)N * 256, stream);
  launch_scatter(bufA, src, tgt, agg, E, 256, stream);
  launch_gemm(agg, invdeg, 256, bufA, 256, W0_l, W0_r, b0, bufB, N, 128, 1, stream);

  // ---- Layer 3: 128 -> 128 ----
  launch_zero(agg, (long long)N * 128, stream);
  launch_scatter(bufB, src, tgt, agg, E, 128, stream);
  launch_gemm(agg, invdeg, 128, bufB, 128, W1_l, W1_r, b1, bufA, N, 128, 1, stream);

  // ---- Head: linear 128 -> 128 (no relu), then 128 -> 8 ----
  launch_gemm(bufA, nullptr, 128, nullptr, 0, W_lin, nullptr, b_lin, bufB, N, 128, 0, stream);
  launch_gemm(bufB, nullptr, 128, nullptr, 0, W_lin2, nullptr, b_lin2, (float*)d_out, N, 8, 0, stream);
}
